// CE_73976516706679
// MI455X (gfx1250) — compile-verified, matrix operand order checked by reading, not run on hardware
//
#include <hip/hip_runtime.h>
#include <hip/hip_bf16.h>

typedef __attribute__((ext_vector_type(16))) _Float16 v16h;
typedef __attribute__((ext_vector_type(8)))  _Float16 v8h;
typedef __attribute__((ext_vector_type(8)))  float    v8f;

#define T_      4
#define CIN     64
#define H_      128
#define W_      128
#define CI_     16
#define PS_     7
#define D_      784      // CI_ * PS_ * PS_
#define DP_     800      // K padded to multiple of 32
#define LDA_    808      // LDS row stride in halves (bank-conflict dodge)
#define S0_     4
#define WS_     27
#define HW_     13
#define KS_     100
#define NROW    27
#define NCOL    80       // candidate-column union window per 16-query half-row
#define SCALE_  10.0f
#define KC_     576      // conv GEMM K = 64ch * 9 taps = 18 * 32
#define XW_     72       // LDS stride (halves) for conv input window
#define LDAW_   584      // LDS stride (halves) for conv weight rows

// ---------------- vid fp32 [t][c][p] -> f16 [t][p][c] (channel-innermost) -----------
__global__ __launch_bounds__(256) void k_tr(const float* __restrict__ vid,
                                            _Float16* __restrict__ vidT)
{
    int gid = blockIdx.x*256 + threadIdx.x;     // t*16384 + p
    if (gid >= T_*H_*W_) return;
    int t = gid >> 14;
    _Float16* dst = vidT + ((size_t)gid << 6);
    const float* src = vid + (((size_t)t*CIN) << 14) + (gid & (H_*W_-1));
    #pragma unroll
    for (int c = 0; c < CIN; ++c)
        dst[c] = (_Float16)src[(size_t)c << 14];
}

// ---------------- reorder conv weights to f16 GEMM-A layout -------------------------
// wgh[o][k], k = tap*64 + c  (tap = ki*3+kj);  wth[o][c]
__global__ void k_prepw(const float* __restrict__ gw, const float* __restrict__ tw,
                        _Float16* __restrict__ wgh, _Float16* __restrict__ wth)
{
    int idx = blockIdx.x*256 + threadIdx.x;
    if (idx < CI_*KC_) {
        int o = idx / KC_, k = idx % KC_;
        int tap = k >> 6, c = k & 63;
        wgh[idx] = (_Float16)gw[(o*CIN + c)*9 + tap];
    }
    if (idx < CI_*CIN)
        wth[idx] = (_Float16)tw[idx];
}

// ---------------- WMMA im2col conv: 3x3 (g) + fused 1x1 (theta) ---------------------
// block = (t, i): one image row; M=16 out-ch, N=16-pixel tiles, K=576
__global__ __launch_bounds__(128) void k_conv_wmma(const _Float16* __restrict__ vidT,
        const _Float16* __restrict__ wgh, const _Float16* __restrict__ wth,
        const float* __restrict__ gb, const float* __restrict__ tb,
        _Float16* __restrict__ v1h, _Float16* __restrict__ v2h)
{
    __shared__ _Float16 X[3*130*XW_];     // rows i-1..i+1, cols -1..128, 64 ch
    __shared__ _Float16 Ag[CI_*LDAW_];
    __shared__ _Float16 At[CI_*XW_];
    int bx = blockIdx.x;                  // t*128 + i
    int t = bx >> 7;
    int i = bx & 127;
    int tid = threadIdx.x;

    for (int u = tid; u < 3*130*8; u += 128) {
        int rr  = u / (130*8);
        int rem = u % (130*8);
        int col = rem >> 3;               // 0..129  (source j = col-1)
        int cc8 = (rem & 7) * 8;
        int ii = i + rr - 1;
        int jj = col - 1;
        v8h val = {};
        if (ii >= 0 && ii < H_ && jj >= 0 && jj < W_)
            val = *(const v8h*)(vidT + ((((size_t)t << 14) + ii*W_ + jj) << 6) + cc8);
        *(v8h*)(&X[(rr*130 + col)*XW_ + cc8]) = val;
    }
    for (int idx = tid; idx < CI_*KC_; idx += 128)
        Ag[(idx/KC_)*LDAW_ + (idx%KC_)] = wgh[idx];
    for (int idx = tid; idx < CI_*CIN; idx += 128)
        At[(idx >> 6)*XW_ + (idx & 63)] = wth[idx];
    __syncthreads();

    int lane = tid & 31;
    int wave = tid >> 5;
    int r0   = lane & 15;                 // A row (out ch) == B pixel lane
    int hi8  = (lane >> 4) * 8;

    for (int nt = wave; nt < 8; nt += 4) {
        int j0 = nt * 16;
        int px = j0 + r0;
        const _Float16* Arow = Ag + r0*LDAW_;
        const _Float16* Atr  = At + r0*XW_;
        v8f acc1 = {};
        v8f acc2 = {};
        for (int s = 0; s < KC_/32; ++s) {
            int k0  = s*32 + hi8;
            int tap = k0 >> 6;
            int c0  = k0 & 63;
            int ki = tap / 3, kj = tap % 3;
            const _Float16* Bp = &X[(ki*130 + px + kj)*XW_ + c0];
            v8h b0 = *(const v8h*)Bp;
            v8h b1 = *(const v8h*)(Bp + 16);
            v8h a0 = *(const v8h*)(Arow + k0);
            v8h a1 = *(const v8h*)(Arow + k0 + 16);
            v16h a, b;
            #pragma unroll
            for (int e = 0; e < 8; ++e) { a[e]=a0[e]; a[e+8]=a1[e]; b[e]=b0[e]; b[e+8]=b1[e]; }
            acc1 = __builtin_amdgcn_wmma_f32_16x16x32_f16(false, a, false, b,
                                                          (short)0, acc1, false, false);
            if (tap == 4) {               // center tap == the 1x1 theta conv's K
                v8h t0 = *(const v8h*)(Atr + c0);
                v8h t1 = *(const v8h*)(Atr + c0 + 16);
                v16h ta;
                #pragma unroll
                for (int e = 0; e < 8; ++e) { ta[e]=t0[e]; ta[e+8]=t1[e]; }
                acc2 = __builtin_amdgcn_wmma_f32_16x16x32_f16(false, ta, false, b,
                                                              (short)0, acc2, false, false);
            }
        }
        int nloc  = lane & 15;
        int mbase = (lane >> 4) * 8;
        #pragma unroll
        for (int jj8 = 0; jj8 < 8; ++jj8) {
            int m = mbase + jj8;
            size_t dst = (((size_t)t*CI_ + m) << 14) + i*W_ + j0 + nloc;
            v1h[dst] = (_Float16)(acc1[jj8] + gb[m]);
            v2h[dst] = (_Float16)(acc2[jj8] + tb[m]);
        }
    }
}

// ---------------- patchify: v[t,16,128,128] f16 -> P[t*N][800] f16 (edge-clamped) ----
__global__ __launch_bounds__(128) void k_patchify(const _Float16* __restrict__ v,
                                                  _Float16* __restrict__ P)
{
    int n = blockIdx.x;                 // t*16384 + i*128 + j
    int t = n >> 14;
    int p = n & (H_*W_ - 1);
    int i = p >> 7, j = p & 127;
    _Float16* dst = P + (size_t)n * DP_;
    const _Float16* src = v + (((size_t)t * CI_) << 14);
    for (int u = threadIdx.x; u < DP_/8; u += 128) {
        v8h chunk;
        #pragma unroll
        for (int e = 0; e < 8; ++e) {
            int d = u*8 + e;
            _Float16 val = (_Float16)0.0f;
            if (d < D_) {
                int c = d / 49, rsd = d % 49;
                int ii = min(i + rsd/7, H_ - 1);
                int jj = min(j + rsd%7, W_ - 1);
                val = src[(c << 14) + ii*W_ + jj];
            }
            chunk[e] = val;
        }
        *(v8h*)(dst + u*8) = chunk;
    }
}

// ---------------- WMMA score GEMM: S = Pq(16x800) * Pc(16x800)^T per tile -----------
__global__ __launch_bounds__(128) void k_scores(const _Float16* __restrict__ P,
                                                float* __restrict__ Sg)
{
    __shared__ _Float16 Aq[16 * LDA_];
    int bx = blockIdx.x;                // t*64 + qi*2 + half
    int t = bx >> 6;
    int qi = (bx >> 1) & 31;
    int half = bx & 1;
    int qi_px = qi * S0_;
    int colbase = half ? 48 : 0;
    int tid = threadIdx.x;

    for (int u = tid; u < 16 * (DP_/8); u += 128) {
        int m  = u / (DP_/8);
        int du = u % (DP_/8);
        int qj_px = (half*16 + m) * S0_;
        const _Float16* src = P + ((size_t)(t*16384) + qi_px*W_ + qj_px) * DP_ + du*8;
        *(v8h*)(&Aq[m*LDA_ + du*8]) = *(const v8h*)src;
    }
    __syncthreads();

    int lane = tid & 31;
    int wave = tid >> 5;
    int r0   = lane & 15;
    int hi8  = (lane >> 4) * 8;
    float* Sblk = Sg + (size_t)bx * (16 * NROW * NCOL);

    for (int tile = wave; tile < NROW*5; tile += 4) {
        int r  = tile / 5;
        int ct = tile % 5;
        int ci = min(max(qi_px + r - HW_, 0), H_ - 1);
        int ccol = colbase + ct*16 + r0;
        const _Float16* Brow = P + ((size_t)(t*16384) + ci*W_ + ccol) * DP_;
        const _Float16* Arow = Aq + r0 * LDA_;
        v8f acc = {};
        for (int s = 0; s < DP_/32; ++s) {
            int kb = s*32 + hi8;             // ISA 16-bit A/B fragment layout (wave32)
            v8h a0 = *(const v8h*)(Arow + kb);
            v8h a1 = *(const v8h*)(Arow + kb + 16);
            v8h b0 = *(const v8h*)(Brow + kb);
            v8h b1 = *(const v8h*)(Brow + kb + 16);
            v16h a, b;
            #pragma unroll
            for (int e = 0; e < 8; ++e) { a[e]=a0[e]; a[e+8]=a1[e]; b[e]=b0[e]; b[e+8]=b1[e]; }
            acc = __builtin_amdgcn_wmma_f32_16x16x32_f16(false, a, false, b,
                                                         (short)0, acc, false, false);
        }
        int nloc  = lane & 15;
        int mbase = (lane >> 4) * 8;
        #pragma unroll
        for (int jj = 0; jj < 8; ++jj)
            Sblk[(mbase + jj) * (NROW*NCOL) + r * NCOL + ct*16 + nloc] = acc[jj];
    }
}

// ---------------- top-100 + softmax: one wave32 per query, register-resident --------
__global__ __launch_bounds__(32) void k_topk(const float* __restrict__ Sg,
        float* __restrict__ wgt, int* __restrict__ inds)
{
    int b = blockIdx.x;                 // t*1024 + qi*32 + qj
    int qj = b & 31;
    int qi = (b >> 5) & 31;
    int half = qj >> 4;
    int m = qj & 15;
    int qi_px = qi * S0_, qj_px = qj * S0_;
    int colbase = half ? 48 : 0;
    const float* Sq = Sg + ((size_t)(((b >> 5) << 1) + half) * 16 + m) * (NROW*NCOL);
    int lane = threadIdx.x;

    const int NL = 23;                  // ceil(729/32)
    float vals[NL];
    unsigned taken = 0;
    #pragma unroll
    for (int u = 0; u < NL; ++u) {
        int s = u*32 + lane;
        float v = -3.4e38f;
        if (s < WS_*WS_) {
            int jc = s % WS_;
            int cj = min(max(qj_px + jc - HW_, 0), W_-1);
            v = Sq[(s / WS_) * NCOL + (cj - colbase)];
        }
        vals[u] = v;
    }

    float ev[4] = {0.f, 0.f, 0.f, 0.f};
    int   nv[4] = {0, 0, 0, 0};
    float vmax = 0.f, esum = 0.f;
    for (int k = 0; k < KS_; ++k) {
        float bv = -3.4e38f; int bi = 0x7fffffff;
        #pragma unroll
        for (int u = 0; u < NL; ++u) {      // register scan, skip taken
            float v = ((taken >> u) & 1u) ? -3.4e38f : vals[u];
            int s = u*32 + lane;
            if (v > bv || (v == bv && s < bi)) { bv = v; bi = s; }
        }
        #pragma unroll
        for (int off = 16; off > 0; off >>= 1) {   // wave32 argmax, tie -> lower index
            float ov = __shfl_xor(bv, off, 32);
            int   oi = __shfl_xor(bi, off, 32);
            if (ov > bv || (ov == bv && oi < bi)) { bv = ov; bi = oi; }
        }
        if ((bi & 31) == lane) taken |= 1u << (bi >> 5);
        if (k == 0) vmax = bv;
        float e = __expf((bv - vmax) * SCALE_);
        esum += e;
        if ((k & 31) == lane) {
            int u = k >> 5;
            ev[u] = e;
            int r = bi / WS_, jc = bi % WS_;
            int ci = min(max(qi_px + r - HW_, 0), H_-1);
            int cj = min(max(qj_px + jc - HW_, 0), W_-1);
            nv[u] = ci*W_ + cj;
        }
    }
    float inv = 1.0f / esum;            // esum identical on all lanes
    #pragma unroll
    for (int u = 0; u < 4; ++u) {
        int k = u*32 + lane;
        if (k < KS_) {
            wgt [(size_t)b*KS_ + k] = ev[u] * inv;
            inds[(size_t)b*KS_ + k] = nv[u];
        }
    }
}

// ---------------- fold normalizer counts --------------------------------------------
__global__ void k_zc(float* __restrict__ Zc) {
    int gid = blockIdx.x*256 + threadIdx.x;
    if (gid >= 1024*49) return;
    int q = gid / 49, s = gid % 49;
    int ii = min((q >> 5) * S0_ + s/7, H_-1);
    int jj = min((q & 31) * S0_ + s%7, W_-1);
    atomicAdd(&Zc[ii*W_ + jj], 1.0f);
}

// ---------------- weighted aggregate + fold (atomic overlap-add) --------------------
__global__ __launch_bounds__(256) void k_agg(const _Float16* __restrict__ P2,
        const float* __restrict__ wgt, const int* __restrict__ inds,
        float* __restrict__ Y)
{
    __shared__ float lw[KS_];
    __shared__ int   ln[KS_];
    int b = blockIdx.x;                 // t*1024 + q
    int t = b >> 10;
    int q = b & 1023;
    int qi = (q >> 5) * S0_, qj = (q & 31) * S0_;
    int tid = threadIdx.x;
    if (tid < KS_) { lw[tid] = wgt[(size_t)b*KS_ + tid]; ln[tid] = inds[(size_t)b*KS_ + tid]; }
    __syncthreads();
    for (int d = tid; d < D_; d += 256) {
        float acc = 0.f;
        for (int k = 0; k < KS_; ++k)
            acc = fmaf(lw[k], (float)P2[((size_t)(t*16384) + ln[k]) * DP_ + d], acc);
        int c = d / 49, rsd = d % 49;
        int ii = min(qi + rsd/7, H_-1);
        int jj = min(qj + rsd%7, W_-1);
        atomicAdd(&Y[(((size_t)t*CI_ + c) << 14) + ii*W_ + jj], acc);
    }
}

__global__ void k_norm(const float* __restrict__ Y, const float* __restrict__ Zc,
                       float* __restrict__ out) {
    int gid = blockIdx.x*256 + threadIdx.x;
    if (gid >= T_*CI_*H_*W_) return;
    out[gid] = Y[gid] / Zc[gid & (H_*W_-1)];
}

extern "C" void kernel_launch(void* const* d_in, const int* in_sizes, int n_in,
                              void* d_out, int out_size, void* d_ws, size_t ws_size,
                              hipStream_t stream)
{
    const float* vid = (const float*)d_in[0];
    const float* gw  = (const float*)d_in[1];
    const float* gb  = (const float*)d_in[2];
    const float* tw  = (const float*)d_in[3];
    const float* tb  = (const float*)d_in[4];
    float* out = (float*)d_out;

    char* w = (char*)d_ws;
    size_t off = 0;
    auto alloc = [&](size_t bytes) {
        void* p = w + off;
        off = (off + bytes + 255) & ~(size_t)255;
        return p;
    };
    _Float16* vidT = (_Float16*)alloc((size_t)T_*H_*W_*CIN*2);      //   8 MB
    _Float16* wgh  = (_Float16*)alloc((size_t)CI_*KC_*2);
    _Float16* wth  = (_Float16*)alloc((size_t)CI_*CIN*2);
    _Float16* v1h  = (_Float16*)alloc((size_t)T_*CI_*H_*W_*2);      //   2 MB
    _Float16* v2h  = (_Float16*)alloc((size_t)T_*CI_*H_*W_*2);      //   2 MB
    _Float16* P    = (_Float16*)alloc((size_t)T_*H_*W_*DP_*2);      // 105 MB (L2-resident)
    float*    Sg   = (float*)   alloc((size_t)T_*64*16*NROW*NCOL*4);//  35 MB
    float*    wgt  = (float*)   alloc((size_t)T_*1024*KS_*4);
    int*      ind  = (int*)     alloc((size_t)T_*1024*KS_*4);
    float*    Y    = (float*)   alloc((size_t)T_*CI_*H_*W_*4);
    float*    Zc   = (float*)   alloc((size_t)H_*W_*4);

    k_tr       <<<(T_*H_*W_+255)/256, 256, 0, stream>>>(vid, vidT);
    k_prepw    <<<(CI_*KC_+255)/256, 256, 0, stream>>>(gw, tw, wgh, wth);
    k_conv_wmma<<<T_*H_, 128, 0, stream>>>(vidT, wgh, wth, gb, tb, v1h, v2h);
    k_patchify <<<T_*H_*W_, 128, 0, stream>>>(v1h, P);     // P = patches(v1)
    k_scores   <<<T_*64, 128, 0, stream>>>(P, Sg);         // WMMA correlation
    k_topk     <<<T_*1024, 32, 0, stream>>>(Sg, wgt, ind);
    k_patchify <<<T_*H_*W_, 128, 0, stream>>>(v2h, P);     // reuse P = patches(v2)
    hipMemsetAsync(Y,  0, (size_t)T_*CI_*H_*W_*4, stream);
    hipMemsetAsync(Zc, 0, (size_t)H_*W_*4, stream);
    k_zc       <<<(1024*49+255)/256, 256, 0, stream>>>(Zc);
    k_agg      <<<T_*1024, 256, 0, stream>>>(P, wgt, ind, Y);
    k_norm     <<<(T_*CI_*H_*W_+255)/256, 256, 0, stream>>>(Y, Zc, out);
}
